// S4Mamba_91036126806279
// MI455X (gfx1250) — matrix operand order repaired and checked by reading, not compiled, not run on hardware
//
#include <hip/hip_runtime.h>

// ---------------------------------------------------------------------------
// MI455X (gfx1250, wave32) implementation of the S4Mamba reference model.
// All dense layers use V_WMMA_F32_16X16X32_F16 (f16 in, fp32 accumulate).
// Sequential SSM scans run in fp32 VALU with 16-lane shfl_xor reductions.
// ---------------------------------------------------------------------------

typedef __attribute__((ext_vector_type(16))) _Float16 v16h;
typedef __attribute__((ext_vector_type(8)))  _Float16 v8h;
typedef __attribute__((ext_vector_type(8)))  float    v8f;

#define CEILDIV(a, b) (((a) + (b) - 1) / (b))

static __device__ __forceinline__ float siluf(float x) { return x / (1.f + __expf(-x)); }

// ---------------------------------------------------------------------------
// WMMA fragment loaders (layouts per CDNA5 ISA 7.12.2, wave32)
// A: row-major M x K.  lane l holds row m = (l&15); elements 0..7 are
//    k = half*8 + 0..7, elements 8..15 are k = 16 + half*8 + 0..7 (half=l>>4)
//    -> two contiguous 16B loads.
// W: stored N x K row-major ("weight (out,in)" layout).  lane l holds col
//    n = (l&15); same k order as A -> two contiguous 16B loads.
// ---------------------------------------------------------------------------
static __device__ __forceinline__ v16h load_a_frag(const _Float16* A, int ldk,
                                                   int m0, int k0, int lane) {
  int m = m0 + (lane & 15);
  int half = lane >> 4;
  const _Float16* p = A + (size_t)m * ldk + k0 + half * 8;
  v16h a;
  *(v8h*)&a       = *(const v8h*)p;
  *((v8h*)&a + 1) = *(const v8h*)(p + 16);
  return a;
}

static __device__ __forceinline__ v16h load_b_frag(const _Float16* W, int ldk,
                                                   int n0, int k0, int lane) {
  int n = n0 + (lane & 15);
  int half = lane >> 4;
  const _Float16* p = W + (size_t)n * ldk + k0 + half * 8;
  v16h b;
  *(v8h*)&b       = *(const v8h*)p;
  *((v8h*)&b + 1) = *(const v8h*)(p + 16);
  return b;
}

// C = act(A(M,K) * W(N,K)^T + bias), optional f32 and/or f16 outputs.
// One 16x16 tile per wave; batched via blockIdx.y with element strides.
// act: 0=none 1=relu 2=silu
__global__ __launch_bounds__(256)
void k_wmma_linear(const _Float16* __restrict__ A, long sA,
                   const _Float16* __restrict__ W, long sW,
                   const float* __restrict__ bias, float* __restrict__ C, long sC,
                   _Float16* __restrict__ C16, long sC16, int M, int N, int K, int act) {
  int wave = (int)((blockIdx.x * blockDim.x + threadIdx.x) >> 5);
  int lane = threadIdx.x & 31;
  int tilesN = N >> 4;
  int nwaves = (M >> 4) * tilesN;
  if (wave >= nwaves) return;                 // wave-uniform: EXEC stays full
  A += (long)blockIdx.y * sA;
  W += (long)blockIdx.y * sW;
  int tm = wave / tilesN, tn = wave % tilesN;
  v8f acc = {};
  for (int k0 = 0; k0 < K; k0 += 32) {
    v16h a = load_a_frag(A, K, tm * 16, k0, lane);
    v16h b = load_b_frag(W, K, tn * 16, k0, lane);
    if (k0 + 32 < K) {                        // WGP-scope global_prefetch_b8
      __builtin_prefetch(A + (size_t)(tm * 16 + (lane & 15)) * K + k0 + 32, 0, 3);
      __builtin_prefetch(W + (size_t)(tn * 16 + (lane & 15)) * K + k0 + 32, 0, 3);
    }
    acc = __builtin_amdgcn_wmma_f32_16x16x32_f16(false, a, false, b,
                                                 (short)0, acc, false, false);
  }
  int col = tn * 16 + (lane & 15);
  float bv = bias ? bias[col] : 0.f;
  int mbase = tm * 16 + (lane >> 4) * 8;
  float* Cb = C ? C + (long)blockIdx.y * sC : nullptr;
  _Float16* C16b = C16 ? C16 + (long)blockIdx.y * sC16 : nullptr;
#pragma unroll
  for (int r = 0; r < 8; ++r) {
    float v = acc[r] + bv;
    if (act == 1) v = fmaxf(v, 0.f);
    else if (act == 2) v = siluf(v);
    size_t idx = (size_t)(mbase + r) * N + col;
    if (Cb)   Cb[idx]   = v;
    if (C16b) C16b[idx] = (_Float16)v;
  }
}

// ---------------------------------------------------------------------------
// Scalar helpers
// ---------------------------------------------------------------------------
__global__ void k_cvt_f16(const float* in, _Float16* out, long n) {
  long i = (long)blockIdx.x * blockDim.x + threadIdx.x;
  if (i < n) out[i] = (_Float16)in[i];
}

__global__ void k_pad_rows_f16(const float* in, _Float16* out, int rin, int rout, int cols) {
  long i = (long)blockIdx.x * blockDim.x + threadIdx.x;
  if (i >= (long)rout * cols) return;
  int r = (int)(i / cols);
  out[i] = (r < rin) ? (_Float16)in[i] : (_Float16)0.f;
}

__global__ void k_pad_vec_f32(const float* in, float* out, int nin, int nout) {
  int i = blockIdx.x * blockDim.x + threadIdx.x;
  if (i < nout) out[i] = (i < nin) ? in[i] : 0.f;
}

__global__ void k_neg_exp(const float* in, float* out, int n) {
  int i = blockIdx.x * blockDim.x + threadIdx.x;
  if (i < n) out[i] = -__expf(in[i]);
}

// act: 0 none, 1 relu, 2 silu, 3 softplus
__global__ void k_linear_naive(const float* A, int lda, const float* W, const float* bias,
                               float* C, int ldc, long M, int N, int K, int act) {
  long i = (long)blockIdx.x * blockDim.x + threadIdx.x;
  if (i >= M * (long)N) return;
  long m = i / N; int n = (int)(i % N);
  const float* a = A + m * lda;
  const float* w = W + (long)n * K;
  float s = bias ? bias[n] : 0.f;
  for (int k = 0; k < K; ++k) s += a[k] * w[k];
  if (act == 1) s = fmaxf(s, 0.f);
  else if (act == 2) s = siluf(s);
  else if (act == 3) s = (s > 20.f) ? s : logf(1.f + __expf(s));
  C[m * ldc + n] = s;
}

// GroupNorm over token-major (B, HW, C); block per (b, group).
// mode bit0 = silu, bit1 = add residual (applied after silu, as in reference)
__global__ __launch_bounds__(256)
void k_groupnorm(const float* x, const float* res, const float* gw, const float* gb,
                 float* out, _Float16* out16, int HW, int C, int G, int mode) {
  int b = blockIdx.x / G, gi = blockIdx.x % G;
  int cpg = C / G;
  long base = (long)b * HW * C;
  int n = HW * cpg;
  __shared__ float s1[256], s2[256];
  float sum = 0.f, sq = 0.f;
  for (int i = threadIdx.x; i < n; i += 256) {
    int pix = i / cpg, c = gi * cpg + i % cpg;
    float v = x[base + (long)pix * C + c];
    sum += v; sq += v * v;
  }
  s1[threadIdx.x] = sum; s2[threadIdx.x] = sq; __syncthreads();
  for (int st = 128; st > 0; st >>= 1) {
    if (threadIdx.x < st) { s1[threadIdx.x] += s1[threadIdx.x + st];
                            s2[threadIdx.x] += s2[threadIdx.x + st]; }
    __syncthreads();
  }
  float mean = s1[0] / n;
  float var  = s2[0] / n - mean * mean;
  float inv  = rsqrtf(var + 1e-5f);
  for (int i = threadIdx.x; i < n; i += 256) {
    int pix = i / cpg, c = gi * cpg + i % cpg;
    long idx = base + (long)pix * C + c;
    float v = (x[idx] - mean) * inv * gw[c] + gb[c];
    if (mode & 1) v = siluf(v);
    if (mode & 2) v += res[idx];
    if (out)   out[idx]   = v;
    if (out16) out16[idx] = (_Float16)v;
  }
}

// im2col for 3x3 SAME conv, (B,HW,64) f32 -> (T, 576) f16, k = i*9+ky*3+kx
__global__ void k_im2col(const float* src, _Float16* col, long total) {
  long i = (long)blockIdx.x * blockDim.x + threadIdx.x;
  if (i >= total) return;
  int k = (int)(i % 576); long m = i / 576;
  int ic = k / 9, r = k % 9, ky = r / 3, kx = r % 3;
  int b = (int)(m / 1024), pix = (int)(m % 1024);
  int h = pix / 32, w = pix % 32;
  int hh = h + ky - 1, ww = w + kx - 1;
  float v = 0.f;
  if (hh >= 0 && hh < 32 && ww >= 0 && ww < 32)
    v = src[((long)b * 1024 + hh * 32 + ww) * 64 + ic];
  col[i] = (_Float16)v;
}

// causal depthwise conv1d (dc=4, left pad 3) + bias + silu
// xin = xz base (row stride ld, channel c at column c)
__global__ void k_causal_conv_silu(const float* xz, int ld, const float* cw, const float* cb,
                                   float* xc, _Float16* xc16, int Bb, int L, int di) {
  long idx = (long)blockIdx.x * blockDim.x + threadIdx.x;
  long total = (long)Bb * L * di;
  if (idx >= total) return;
  int c = (int)(idx % di); long bl = idx / di;
  int l = (int)(bl % L); long b = bl / L;
  const float* base = xz + ((long)b * L) * ld + c;
  float s = cb[c];
#pragma unroll
  for (int k = 0; k < 4; ++k) {
    int ls = l + k - 3;
    if (ls >= 0) s += cw[c * 4 + k] * base[(long)ls * ld];
  }
  s = siluf(s);
  xc[idx] = s;
  if (xc16) xc16[idx] = (_Float16)s;
}

// selective scan: thread=(b,d,s), ds=16, sum over s via 16-lane shfl_xor
__global__ __launch_bounds__(256)
void k_mamba_scan(const float* delta, const float* BC, const float* xc, const float* A,
                  float* y, int Bb, int L, int di, int ldb, int boff, int coff) {
  int t = threadIdx.x;
  int s = t & 15, dloc = t >> 4;                 // 16 channels x 16 states
  int chunks = di / 16;
  long gid = blockIdx.x;
  long b = gid / chunks;
  int d = (int)(gid % chunks) * 16 + dloc;
  float a = A[d * 16 + s];
  float h = 0.f;
  const float* dp = delta + (b * L) * di + d;
  const float* xp = xc    + (b * L) * di + d;
  const float* bp = BC + (b * L) * ldb + boff + s;
  const float* cp = BC + (b * L) * ldb + coff + s;
  float* yp = y + (b * L) * di + d;
  for (int l = 0; l < L; ++l) {
    float dt = *dp, Bv = *bp, Cv = *cp, xv = *xp;
    h = __expf(dt * a) * h + dt * Bv * xv;
    float pr = h * Cv;
    pr += __shfl_xor(pr, 1, 16);
    pr += __shfl_xor(pr, 2, 16);
    pr += __shfl_xor(pr, 4, 16);
    pr += __shfl_xor(pr, 8, 16);
    if (s == 0) *yp = pr;
    dp += di; xp += di; bp += ldb; cp += ldb; yp += di;
  }
}

// y = (yscan + xc*D) * silu(z),  z at xz[:, zoff + d] (row stride ldz)
__global__ void k_gate(const float* ysc, const float* xc, const float* xz, const float* Dv,
                       float* out32, _Float16* out16, long M, int di, int ldz, int zoff) {
  long idx = (long)blockIdx.x * blockDim.x + threadIdx.x;
  if (idx >= M * (long)di) return;
  long m = idx / di; int d = (int)(idx % di);
  float z = xz[m * ldz + zoff + d];
  float v = (ysc[idx] + xc[idx] * Dv[d]) * siluf(z);
  if (out32) out32[idx] = v;
  if (out16) out16[idx] = (_Float16)v;
}

// -------------------- similarity-module kernels ----------------------------
__global__ void k_entropy(const float* ab, float* ent, long T, int ld, int Cc) {
  long m = (long)blockIdx.x * blockDim.x + threadIdx.x;
  if (m >= T) return;
  const float* a = ab + m * ld;
  float mx = -1e30f;
  for (int c = 0; c < Cc; ++c) mx = fmaxf(mx, a[c]);
  float sum = 0.f;
  for (int c = 0; c < Cc; ++c) sum += __expf(a[c] - mx);
  float inv = 1.f / sum, H = 0.f;
  for (int c = 0; c < Cc; ++c) {
    float p = __expf(a[c] - mx) * inv;
    H -= p * logf(p + 1e-8f);
  }
  ent[m] = H;
}

__global__ __launch_bounds__(256)
void k_argmax_sel(const float* ab, const float* ent, float* sel, float* selnorm,
                  int Np, int Cc, int ld) {
  int b = blockIdx.x;
  __shared__ float sv[256];
  __shared__ int   si[256];
  float best = -1e30f; int bi = 0;
  for (int n = threadIdx.x; n < Np; n += 256) {
    float v = ent[(long)b * Np + n];
    if (v > best) { best = v; bi = n; }
  }
  sv[threadIdx.x] = best; si[threadIdx.x] = bi; __syncthreads();
  for (int st = 128; st > 0; st >>= 1) {
    if (threadIdx.x < st) {
      float v2 = sv[threadIdx.x + st]; int i2 = si[threadIdx.x + st];
      if (v2 > sv[threadIdx.x] || (v2 == sv[threadIdx.x] && i2 < si[threadIdx.x])) {
        sv[threadIdx.x] = v2; si[threadIdx.x] = i2;
      }
    }
    __syncthreads();
  }
  int idx = si[0]; __syncthreads();
  float sq = 0.f;
  if (threadIdx.x < Cc) {
    float v = ab[((long)b * Np + idx) * ld + threadIdx.x];
    sel[b * Cc + threadIdx.x] = v;
    sq = v * v;
  }
  sv[threadIdx.x] = sq; __syncthreads();
  for (int st = 128; st > 0; st >>= 1) {
    if (threadIdx.x < st) sv[threadIdx.x] += sv[threadIdx.x + st];
    __syncthreads();
  }
  if (threadIdx.x == 0) selnorm[b] = fmaxf(sqrtf(sv[0]), 1e-8f);
}

__global__ void k_simv(const float* ab, const float* sel, const float* selnorm,
                       float* sv, long T, int Np, int Cc, int ld) {
  long m = (long)blockIdx.x * blockDim.x + threadIdx.x;
  if (m >= T) return;
  int b = (int)(m / Np);
  const float* bsp = ab + m * ld + Cc;
  float num = 0.f, nb = 0.f;
  for (int c = 0; c < Cc; ++c) {
    float bv = bsp[c];
    num += sel[b * Cc + c] * bv;
    nb  += bv * bv;
  }
  sv[m] = num / (selnorm[b] * fmaxf(sqrtf(nb), 1e-8f));
}

__global__ __launch_bounds__(256)
void k_softmax_sq(const float* sv, float* attn, int Np) {
  int b = blockIdx.x;
  const float* x = sv + (long)b * Np;
  __shared__ float s[256];
  float mx = -1e30f;
  for (int n = threadIdx.x; n < Np; n += 256) { float v = x[n] * x[n]; mx = fmaxf(mx, v); }
  s[threadIdx.x] = mx; __syncthreads();
  for (int st = 128; st > 0; st >>= 1) {
    if (threadIdx.x < st) s[threadIdx.x] = fmaxf(s[threadIdx.x], s[threadIdx.x + st]);
    __syncthreads();
  }
  float M = s[0]; __syncthreads();
  float sum = 0.f;
  for (int n = threadIdx.x; n < Np; n += 256) sum += __expf(x[n] * x[n] - M);
  s[threadIdx.x] = sum; __syncthreads();
  for (int st = 128; st > 0; st >>= 1) {
    if (threadIdx.x < st) s[threadIdx.x] += s[threadIdx.x + st];
    __syncthreads();
  }
  float S = s[0];
  for (int n = threadIdx.x; n < Np; n += 256)
    attn[(long)b * Np + n] = __expf(x[n] * x[n] - M) / S;
}

__global__ void k_spa_x(const float* ab, const float* attn, const float* h2,
                        float* out, _Float16* out16, long total, int Cc, int ld) {
  long i = (long)blockIdx.x * blockDim.x + threadIdx.x;
  if (i >= total) return;
  long m = i / Cc; int c = (int)(i % Cc);
  float v = attn[m] * ab[m * ld + Cc + c] + h2[i];
  out[i] = v; out16[i] = (_Float16)v;
}

// split first 64 cols of ab -> token-major f16 + channel-major f16
__global__ void k_split_a(const float* ab, _Float16* a_tok, _Float16* a_ch,
                          long total, int Cc, int Np, int ld) {
  long i = (long)blockIdx.x * blockDim.x + threadIdx.x;
  if (i >= total) return;
  long m = i / Cc; int c = (int)(i % Cc);
  int b = (int)(m / Np), n = (int)(m % Np);
  float v = ab[m * ld + c];
  a_tok[i] = (_Float16)v;
  a_ch[((long)b * Cc + c) * Np + n] = (_Float16)v;
}

// row softmax of gram/(n_c n_d + eps); block per (b,c), 64 threads
__global__ __launch_bounds__(64)
void k_atten_spe(const float* gram, const float* eps, _Float16* out, int Cc) {
  int b = blockIdx.x / Cc, c = blockIdx.x % Cc;
  int d = threadIdx.x;
  const float* G = gram + (long)b * Cc * Cc;
  float e = eps[0];
  float nc = sqrtf(fmaxf(G[c * Cc + c], 0.f));
  float nd = sqrtf(fmaxf(G[d * Cc + d], 0.f));
  float v = G[c * Cc + d] / (nc * nd + e);
  __shared__ float s[64];
  s[d] = v; __syncthreads();
  for (int st = 32; st > 0; st >>= 1) {
    if (d < st) s[d] = fmaxf(s[d], s[d + st]);
    __syncthreads();
  }
  float mx = s[0]; __syncthreads();
  float ev = __expf(v - mx);
  s[d] = ev; __syncthreads();
  for (int st = 32; st > 0; st >>= 1) {
    if (d < st) s[d] += s[d + st];
    __syncthreads();
  }
  out[((long)b * Cc + c) * Cc + d] = (_Float16)(ev / s[0]);
}

__global__ void k_fusion(const float* spa, const float* spe, const float* attw,
                         const float* saw, const float* sab,
                         const float* sew, const float* seb,
                         _Float16* out, long total, int Cc) {
  long i = (long)blockIdx.x * blockDim.x + threadIdx.x;
  if (i >= total) return;
  int c = (int)(i % Cc);
  float a0 = attw[0], a1 = attw[1];
  float mx = fmaxf(a0, a1);
  float e0 = __expf(a0 - mx), e1 = __expf(a1 - mx);
  float w0 = e0 / (e0 + e1), w1 = e1 / (e0 + e1);
  float t = spa[i] * saw[c] + sab[c]; t = (t >= 0.f) ? t : 0.01f * t;
  float u = spe[i] * sew[c] + seb[c]; u = (u >= 0.f) ? u : 0.01f * u;
  out[i] = (_Float16)(t * w0 + u * w1);
}

__global__ __launch_bounds__(256)
void k_mean_head(const float* o2, float* out, int HW, int ldo, int ncls) {
  int b = blockIdx.x / ncls, j = blockIdx.x % ncls;
  __shared__ float s[256];
  float a = 0.f;
  for (int n = threadIdx.x; n < HW; n += 256)
    a += o2[((long)b * HW + n) * ldo + j];
  s[threadIdx.x] = a; __syncthreads();
  for (int st = 128; st > 0; st >>= 1) {
    if (threadIdx.x < st) s[threadIdx.x] += s[threadIdx.x + st];
    __syncthreads();
  }
  if (threadIdx.x == 0) out[b * ncls + j] = s[0] / HW;
}

// ---------------------------------------------------------------------------
// Host side
// ---------------------------------------------------------------------------
static void wmma_lin(hipStream_t st, const _Float16* A, long sA,
                     const _Float16* W, long sW, const float* bias,
                     float* C, long sC, _Float16* C16, long sC16,
                     int M, int N, int K, int act, int batch) {
  long waves = (long)(M / 16) * (N / 16);
  long thr = waves * 32;
  dim3 grid((unsigned)CEILDIV(thr, 256), (unsigned)batch);
  k_wmma_linear<<<grid, 256, 0, st>>>(A, sA, W, sW, bias, C, sC, C16, sC16, M, N, K, act);
}

extern "C" void kernel_launch(void* const* d_in, const int* in_sizes, int n_in,
                              void* d_out, int out_size, void* d_ws, size_t ws_size,
                              hipStream_t stream) {
  (void)in_sizes; (void)n_in; (void)out_size; (void)ws_size;
  const long T = 32768;          // B*H*W
  const long M2 = 262144;        // (B*H*W) * TOKN rows for spectral mamba
  // ---- inputs (setup_inputs dict order, depth-first) ----
  const float* x        = (const float*)d_in[0];
  const float* pe_w1    = (const float*)d_in[1];
  const float* pe_b1    = (const float*)d_in[2];
  const float* pe_g1    = (const float*)d_in[3];
  const float* pe_be1   = (const float*)d_in[4];
  const float* pe_w2    = (const float*)d_in[5];
  const float* pe_b2    = (const float*)d_in[6];
  const float* sim_w    = (const float*)d_in[7];
  const float* sim_b    = (const float*)d_in[8];
  const float* sim_eps  = (const float*)d_in[9];
  const float* mlp_w1   = (const float*)d_in[10];
  const float* mlp_b1   = (const float*)d_in[11];
  const float* mlp_w2   = (const float*)d_in[12];
  const float* mlp_b2   = (const float*)d_in[13];
  const float* m_inw    = (const float*)d_in[14];
  const float* m_cw     = (const float*)d_in[15];
  const float* m_cb     = (const float*)d_in[16];
  const float* m_xpw    = (const float*)d_in[17];
  const float* m_dtw    = (const float*)d_in[18];
  const float* m_dtb    = (const float*)d_in[19];
  const float* m_Alog   = (const float*)d_in[20];
  const float* m_D      = (const float*)d_in[21];
  const float* m_outw   = (const float*)d_in[22];
  const float* spa_g    = (const float*)d_in[23];
  const float* spa_be   = (const float*)d_in[24];
  const float* s_inw    = (const float*)d_in[25];
  const float* s_cw     = (const float*)d_in[26];
  const float* s_cb     = (const float*)d_in[27];
  const float* s_xpw    = (const float*)d_in[28];
  const float* s_dtw    = (const float*)d_in[29];
  const float* s_dtb    = (const float*)d_in[30];
  const float* s_Alog   = (const float*)d_in[31];
  const float* s_D      = (const float*)d_in[32];
  const float* s_outw   = (const float*)d_in[33];
  const float* spe_g    = (const float*)d_in[34];
  const float* spe_be   = (const float*)d_in[35];
  const float* att_w    = (const float*)d_in[36];
  const float* bo_saw   = (const float*)d_in[37];
  const float* bo_sab   = (const float*)d_in[38];
  const float* bo_sew   = (const float*)d_in[39];
  const float* bo_seb   = (const float*)d_in[40];
  const float* cls_w1   = (const float*)d_in[41];
  const float* cls_b1   = (const float*)d_in[42];
  const float* cls_g    = (const float*)d_in[43];
  const float* cls_be   = (const float*)d_in[44];
  const float* cls_w2   = (const float*)d_in[45];
  const float* cls_b2   = (const float*)d_in[46];
  float* out = (float*)d_out;

  // ---- workspace bump allocator (assumes generously sized d_ws) ----
  size_t off = 0;
  auto alloc = [&](size_t bytes) -> void* {
    void* r = (char*)d_ws + off;
    off += (bytes + 255) & ~(size_t)255;
    return r;
  };
  // f16 weights
  _Float16* w16_pe1  = (_Float16*)alloc(8192 * 2);
  _Float16* w16_pe2  = (_Float16*)alloc(36864 * 2);
  _Float16* w16_sim  = (_Float16*)alloc(8192 * 2);
  _Float16* w16_ml1  = (_Float16*)alloc(8192 * 2);
  _Float16* w16_ml2  = (_Float16*)alloc(8192 * 2);
  _Float16* w16_inp  = (_Float16*)alloc(16384 * 2);
  _Float16* w16_xpp  = (_Float16*)alloc(48 * 128 * 2);   // padded 36->48 rows
  _Float16* w16_outp = (_Float16*)alloc(8192 * 2);
  _Float16* w16_cl1  = (_Float16*)alloc(4096 * 2);
  _Float16* w16_cl2  = (_Float16*)alloc(16 * 64 * 2);    // padded 10->16 rows
  float* clsb2_pad   = (float*)alloc(16 * 4);
  float* A_spa       = (float*)alloc(2048 * 4);
  float* A_spe       = (float*)alloc(256 * 4);
  // activations
  _Float16* x16      = (_Float16*)alloc(T * 128 * 2);
  float* h0          = (float*)alloc(T * 64 * 4);
  float* hgn         = (float*)alloc(T * 64 * 4);
  _Float16* col16    = (_Float16*)alloc(T * 576 * 2);
  float* h2          = (float*)alloc(T * 64 * 4);
  _Float16* h2_16    = (_Float16*)alloc(T * 64 * 2);
  float* ab          = (float*)alloc(T * 128 * 4);
  _Float16* a_tok16  = (_Float16*)alloc(T * 64 * 2);
  _Float16* a_ch16   = (_Float16*)alloc(T * 64 * 2);
  float* ent_b       = (float*)alloc(T * 4);
  float* simv_b      = (float*)alloc(T * 4);
  float* attn_b      = (float*)alloc(T * 4);
  float* sel_b       = (float*)alloc(32 * 64 * 4);
  float* selnorm_b   = (float*)alloc(32 * 4);
  float* gram        = (float*)alloc(32 * 4096 * 4);
  _Float16* attnsp16 = (_Float16*)alloc(32 * 4096 * 2);
  float* spa_x       = (float*)alloc(T * 64 * 4);
  _Float16* spa_x16  = (_Float16*)alloc(T * 64 * 2);
  float* spe_x       = (float*)alloc(T * 64 * 4);
  _Float16* h1_16    = (_Float16*)alloc(T * 128 * 2);
  _Float16* h2m_16   = (_Float16*)alloc(T * 64 * 2);
  float* xz          = (float*)alloc(T * 256 * 4);
  float* xc          = (float*)alloc(T * 128 * 4);
  _Float16* xc16     = (_Float16*)alloc(T * 128 * 2);
  float* dbl         = (float*)alloc(T * 48 * 4);
  float* delta       = (float*)alloc(T * 128 * 4);
  float* yscan       = (float*)alloc(T * 128 * 4);
  _Float16* ytmp16   = (_Float16*)alloc(T * 128 * 2);
  float* mamba_out   = (float*)alloc(T * 64 * 4);
  float* spa_final   = (float*)alloc(T * 64 * 4);
  float* xz2         = (float*)alloc(M2 * 32 * 4);
  float* xc2         = (float*)alloc(M2 * 16 * 4);
  float* dbl2        = (float*)alloc(M2 * 33 * 4);
  float* delta2      = (float*)alloc(M2 * 16 * 4);
  float* yscan2      = (float*)alloc(M2 * 16 * 4);
  float* ytmp2       = (float*)alloc(M2 * 16 * 4);
  float* spe_mout    = (float*)alloc(M2 * 8 * 4);
  float* spe_final   = (float*)alloc(T * 64 * 4);
  _Float16* fus16    = (_Float16*)alloc(T * 64 * 2);
  float* o1          = (float*)alloc(T * 64 * 4);
  _Float16* o1g16    = (_Float16*)alloc(T * 64 * 2);
  float* o2          = (float*)alloc(T * 16 * 4);

  // ---- weight prep ----
  auto cvt = [&](const float* src, _Float16* dst, long n) {
    k_cvt_f16<<<CEILDIV(n, 256), 256, 0, stream>>>(src, dst, n);
  };
  cvt(pe_w1, w16_pe1, 8192);   cvt(pe_w2, w16_pe2, 36864);
  cvt(sim_w, w16_sim, 8192);   cvt(mlp_w1, w16_ml1, 8192);
  cvt(mlp_w2, w16_ml2, 8192);  cvt(m_inw, w16_inp, 16384);
  cvt(m_outw, w16_outp, 8192); cvt(cls_w1, w16_cl1, 4096);
  k_pad_rows_f16<<<CEILDIV(48 * 128, 256), 256, 0, stream>>>(m_xpw, w16_xpp, 36, 48, 128);
  k_pad_rows_f16<<<CEILDIV(16 * 64, 256), 256, 0, stream>>>(cls_w2, w16_cl2, 10, 16, 64);
  k_pad_vec_f32<<<1, 64, 0, stream>>>(cls_b2, clsb2_pad, 10, 16);
  k_neg_exp<<<CEILDIV(2048, 256), 256, 0, stream>>>(m_Alog, A_spa, 2048);
  k_neg_exp<<<1, 256, 0, stream>>>(s_Alog, A_spe, 256);
  cvt(x, x16, T * 128);

  // ---- patch embed: conv1x1 + GN + conv3x3(+bias,silu) ----
  wmma_lin(stream, x16, 0, w16_pe1, 0, pe_b1, h0, 0, nullptr, 0, (int)T, 64, 128, 0, 1);
  k_groupnorm<<<32 * 4, 256, 0, stream>>>(h0, nullptr, pe_g1, pe_be1, hgn, nullptr, 1024, 64, 4, 0);
  k_im2col<<<CEILDIV(T * 576, 256), 256, 0, stream>>>(hgn, col16, T * 576);
  wmma_lin(stream, col16, 0, w16_pe2, 0, pe_b2, h2, 0, h2_16, 0, (int)T, 64, 576, 2, 1);

  // ---- similarity ----
  wmma_lin(stream, h2_16, 0, w16_sim, 0, sim_b, ab, 0, nullptr, 0, (int)T, 128, 64, 0, 1);
  k_entropy<<<CEILDIV(T, 256), 256, 0, stream>>>(ab, ent_b, T, 128, 64);
  k_argmax_sel<<<32, 256, 0, stream>>>(ab, ent_b, sel_b, selnorm_b, 1024, 64, 128);
  k_simv<<<CEILDIV(T, 256), 256, 0, stream>>>(ab, sel_b, selnorm_b, simv_b, T, 1024, 64, 128);
  k_softmax_sq<<<32, 256, 0, stream>>>(simv_b, attn_b, 1024);
  k_spa_x<<<CEILDIV(T * 64, 256), 256, 0, stream>>>(ab, attn_b, h2, spa_x, spa_x16, T * 64, 64, 128);
  k_split_a<<<CEILDIV(T * 64, 256), 256, 0, stream>>>(ab, a_tok16, a_ch16, T * 64, 64, 1024, 128);
  // per-batch channel Gram (64x64, K=1024) and spectral attention apply
  wmma_lin(stream, a_ch16, 64 * 1024, a_ch16, 64 * 1024, nullptr,
           gram, 64 * 64, nullptr, 0, 64, 64, 1024, 0, 32);
  k_atten_spe<<<32 * 64, 64, 0, stream>>>(gram, sim_eps, attnsp16, 64);
  wmma_lin(stream, a_tok16, 1024 * 64, attnsp16, 64 * 64, nullptr,
           spe_x, 1024 * 64, nullptr, 0, 1024, 64, 64, 0, 32);

  // ---- spatial mamba branch ----
  wmma_lin(stream, spa_x16, 0, w16_ml1, 0, mlp_b1, nullptr, 0, h1_16, 0, (int)T, 128, 64, 1, 1);
  wmma_lin(stream, h1_16, 0, w16_ml2, 0, mlp_b2, nullptr, 0, h2m_16, 0, (int)T, 64, 128, 0, 1);
  wmma_lin(stream, h2m_16, 0, w16_inp, 0, nullptr, xz, 0, nullptr, 0, (int)T, 256, 64, 0, 1);
  k_causal_conv_silu<<<CEILDIV(T * 128, 256), 256, 0, stream>>>(xz, 256, m_cw, m_cb, xc, xc16, 32, 1024, 128);
  wmma_lin(stream, xc16, 0, w16_xpp, 0, nullptr, dbl, 0, nullptr, 0, (int)T, 48, 128, 0, 1);
  k_linear_naive<<<CEILDIV(T * 128, 256), 256, 0, stream>>>(dbl, 48, m_dtw, m_dtb, delta, 128, T, 128, 4, 3);
  k_mamba_scan<<<32 * 8, 256, 0, stream>>>(delta, dbl, xc, A_spa, yscan, 32, 1024, 128, 48, 4, 20);
  k_gate<<<CEILDIV(T * 128, 256), 256, 0, stream>>>(yscan, xc, xz, m_D, nullptr, ytmp16, T, 128, 256, 128);
  wmma_lin(stream, ytmp16, 0, w16_outp, 0, nullptr, mamba_out, 0, nullptr, 0, (int)T, 64, 128, 0, 1);
  k_groupnorm<<<32 * 4, 256, 0, stream>>>(mamba_out, spa_x, spa_g, spa_be, spa_final, nullptr, 1024, 64, 4, 3);

  // ---- spectral mamba branch (tiny K -> scalar) ----
  k_linear_naive<<<CEILDIV(M2 * 32, 256), 256, 0, stream>>>(spe_x, 8, s_inw, nullptr, xz2, 32, M2, 32, 8, 0);
  k_causal_conv_silu<<<CEILDIV(M2 * 16, 256), 256, 0, stream>>>(xz2, 32, s_cw, s_cb, xc2, nullptr, 32768, 8, 16);
  k_linear_naive<<<CEILDIV(M2 * 33, 256), 256, 0, stream>>>(xc2, 16, s_xpw, nullptr, dbl2, 33, M2, 33, 16, 0);
  k_linear_naive<<<CEILDIV(M2 * 16, 256), 256, 0, stream>>>(dbl2, 33, s_dtw, s_dtb, delta2, 16, M2, 16, 1, 3);
  k_mamba_scan<<<32768, 256, 0, stream>>>(delta2, dbl2, xc2, A_spe, yscan2, 32768, 8, 16, 33, 1, 17);
  k_gate<<<CEILDIV(M2 * 16, 256), 256, 0, stream>>>(yscan2, xc2, xz2, s_D, ytmp2, nullptr, M2, 16, 32, 16);
  k_linear_naive<<<CEILDIV(M2 * 8, 256), 256, 0, stream>>>(ytmp2, 16, s_outw, nullptr, spe_mout, 8, M2, 8, 16, 0);
  k_groupnorm<<<32 * 4, 256, 0, stream>>>(spe_mout, spe_x, spe_g, spe_be, spe_final, nullptr, 1024, 64, 4, 3);

  // ---- fusion + classifier head ----
  k_fusion<<<CEILDIV(T * 64, 256), 256, 0, stream>>>(spa_final, spe_final, att_w,
      bo_saw, bo_sab, bo_sew, bo_seb, fus16, T * 64, 64);
  wmma_lin(stream, fus16, 0, w16_cl1, 0, cls_b1, o1, 0, nullptr, 0, (int)T, 64, 64, 0, 1);
  k_groupnorm<<<32 * 4, 256, 0, stream>>>(o1, nullptr, cls_g, cls_be, nullptr, o1g16, 1024, 64, 4, 1);
  wmma_lin(stream, o1g16, 0, w16_cl2, 0, clsb2_pad, o2, 0, nullptr, 0, (int)T, 16, 64, 0, 1);
  k_mean_head<<<32 * 10, 256, 0, stream>>>(o2, out, 1024, 16, 10);
}
